// Leaf_14336600834127
// MI455X (gfx1250) — matrix operand order repaired and checked by reading, not compiled, not run on hardware
//
#include <hip/hip_runtime.h>
#include <math.h>

// ---------------------------------------------------------------------------
// LEAF-style frontend for MI455X (gfx1250).
// Dominant cost: 128x3073-tap conv over 4x80000 samples = ~252 GFLOP,
// compute-bound (all data << 192MB L2). Mapped onto v_wmma_f32_16x16x32_f16
// as a Toeplitz GEMM (M=filters(128), K=taps(3073->3104), N=time).
// Pooling (stride 480, 3073-tap Gaussian) fused into the conv epilogue via
// LDS partial sums + global f32 atomics into a 171KB pooled buffer, avoiding
// an 82MB |y|^2 intermediate. Final PCEN/EMA scan: 256 independent rows.
// ---------------------------------------------------------------------------

typedef __attribute__((ext_vector_type(16))) _Float16 v16h;
typedef __attribute__((ext_vector_type(8)))  float    v8f;

#define NB      4
#define T_IN    80000
#define NF      64
#define FS      3073        // FILTER_SIZE == POOL_SIZE
#define PAD     1536        // FS/2
#define NKC     97          // ceil(3073/32) K-chunks
#define KPAD    (NKC*32)    // 3104
#define PSTR    480
#define T_OUTN  167
#define XPS     83136       // padded signal stride >= T_IN + KPAD + 15, 64-aligned
#define EPS_V   1e-12f
#define CLAMP_V 1e-5f

// --------------------------- builder kernels -------------------------------

// Zero-padded f16 copy of the input signal: xp[b][i] = x[b][i-PAD] (0 outside).
__global__ void k_build_xp(const float* __restrict__ x, _Float16* __restrict__ xp) {
  int e = blockIdx.x * 256 + threadIdx.x;
  if (e >= NB * XPS) return;
  int b = e / XPS, i = e % XPS, j = i - PAD;
  float v = (j >= 0 && j < T_IN) ? x[b * T_IN + j] : 0.0f;
  xp[e] = (_Float16)v;
}

// Gabor filters, stored pre-swizzled in the WMMA 16x32 f16 A-fragment layout:
// aswz[mtile(8)][kchunk(97)][lane(32)][h(16)] so the conv kernel does one
// aligned 32B load per lane per fragment.
// A layout (contiguous K): lanes 0-15 -> M=lane, K = h;
//                        lanes 16-31 -> M=lane-16, K = h + 16.
__global__ void k_build_filt(const float* __restrict__ cf_in,
                             const float* __restrict__ bw_in,
                             _Float16* __restrict__ aswz) {
  int e = blockIdx.x * 256 + threadIdx.x;
  if (e >= 8 * NKC * 32 * 16) return;
  int h    = e & 15;
  int lane = (e >> 4) & 31;
  int kc   = (e >> 9) % NKC;
  int mt   = e / (512 * NKC);
  int M    = mt * 16 + (lane & 15);
  int klo  = h;
  int K    = kc * 32 + klo + ((lane >= 16) ? 16 : 0);
  float val = 0.0f;
  if (K < FS) {
    const float zc = 0.3748022f;                    // sqrt(2 ln2)/pi
    int   f   = M & 63;
    float cfv = fminf(fmaxf(cf_in[f], 0.0f), 3.14159265f);
    float bw  = fminf(fmaxf(bw_in[f], 4.0f * zc), (float)FS * zc);
    float t   = (float)(K - PAD);
    float env = (0.3989423f / bw) * expf(-t * t / (2.0f * bw * bw));
    float ph  = cfv * t;
    val = env * ((M < 64) ? cosf(ph) : sinf(ph));   // rows 0-63 re, 64-127 im
  }
  aswz[e] = (_Float16)val;
}

// Gaussian pooling windows, f32, [NF][FS].
__global__ void k_build_win(const float* __restrict__ pw_in, float* __restrict__ win) {
  int e = blockIdx.x * 256 + threadIdx.x;
  if (e >= NF * FS) return;
  int f = e / FS, u = e % FS;
  float pw  = fminf(fmaxf(pw_in[f], 2.0f / (float)FS), 0.5f);
  float num = (float)u * (2.0f / (float)(FS - 1)) - 1.0f;
  float q   = num / pw;
  win[e] = expf(-0.5f * q * q);
}

__global__ void k_zero(float* __restrict__ p, int n) {
  int e = blockIdx.x * 256 + threadIdx.x;
  if (e < n) p[e] = 0.0f;
}

// --------------------------- conv + fused pooling --------------------------
// Block = 256 threads (8 waves). Wave w owns filter M-tile [16w,16w+16);
// block owns one batch b and one 16-wide time tile t0. 97 WMMA steps over K.
// Epilogue: |y|^2 across re/im wave pairs via LDS, then scatter into the
// (<=8) pooled outputs this 16-sample window overlaps.
__global__ __launch_bounds__(256) void k_conv_pool(
    const _Float16* __restrict__ xp, const _Float16* __restrict__ aswz,
    const float* __restrict__ win, float* __restrict__ pooled) {
  __shared__ float ybuf[128 * 17];   // stride 17 to spread banks
  __shared__ float pacc[64 * 8];     // per-block pooled partials

  const int tid  = threadIdx.x;
  const int w    = tid >> 5;
  const int lane = tid & 31;
  const int t0   = blockIdx.x * 16;
  const int b    = blockIdx.y;

  const _Float16* ap = aswz + ((size_t)w * NKC * 32 + lane) * 16;      // 32B aligned
  const _Float16* bp = xp + (size_t)b * XPS + t0 + lane;               // unaligned ok

  v8f c = {};
  #pragma unroll 2
  for (int kc = 0; kc < NKC; ++kc) {
    v16h a = *(const v16h*)ap;          // aligned 32B fragment
    v16h bb;
    __builtin_memcpy(&bb, bp, 32);      // B[k][n] = xp[t0 + n + k]: contiguous per lane
    c = __builtin_amdgcn_wmma_f32_16x16x32_f16(false, a, false, bb,
                                               (short)0, c, false, false);
    ap += 32 * 16;
    bp += 32;
  }

  // C layout: VGPR v, lanes 0-15 -> M=v, N=lane; lanes 16-31 -> M=v+8, N=lane-16
  {
    const int half = lane >> 4, n = lane & 15;
    const int mbase = w * 16 + half * 8;
    #pragma unroll
    for (int v = 0; v < 8; ++v) ybuf[(mbase + v) * 17 + n] = c[v];
  }
  pacc[tid] = 0.0f;
  pacc[tid + 256] = 0.0f;
  __syncthreads();

  // |y|^2 = re^2 + im^2; 4 (f,n) values per thread
  float pv[4]; int pf[4], pn[4];
  #pragma unroll
  for (int j = 0; j < 4; ++j) {
    int idx = tid * 4 + j;
    int f = idx >> 4, n = idx & 15;
    float re = ybuf[f * 17 + n];
    float im = ybuf[(f + 64) * 17 + n];
    pv[j] = re * re + im * im;
    pf[j] = f; pn[j] = n;
  }

  // pooled[to] += win[f][t + PAD - 480*to] * |y|^2[t],  0 <= u < FS
  int num   = t0 - PAD;
  int to_lo = (num <= 0) ? 0 : (num + PSTR - 1) / PSTR;
  int to_hi = (t0 + 15 + PAD) / PSTR;
  if (to_hi > T_OUTN - 1) to_hi = T_OUTN - 1;

  #pragma unroll
  for (int j = 0; j < 4; ++j) {
    for (int to = to_lo; to <= to_hi; ++to) {
      int u = t0 + pn[j] + PAD - to * PSTR;
      if (u >= 0 && u < FS)
        atomicAdd(&pacc[pf[j] * 8 + (to - to_lo)], pv[j] * win[pf[j] * FS + u]);
    }
  }
  __syncthreads();

  #pragma unroll
  for (int j = 0; j < 2; ++j) {
    int idx = tid + 256 * j;
    int f = idx >> 3, slot = idx & 7, to = to_lo + slot;
    if (to <= to_hi) {
      float v = pacc[idx];
      if (v != 0.0f)
        unsafeAtomicAdd(&pooled[((size_t)b * NF + f) * T_OUTN + to], v);
    }
  }
}

// --------------------------- EMA scan + PCEN -------------------------------
__global__ void k_pcen(const float* __restrict__ pooled,
                       const float* __restrict__ s_in, const float* __restrict__ a_in,
                       const float* __restrict__ d_in, const float* __restrict__ r_in,
                       float* __restrict__ out) {
  int tid = threadIdx.x;                 // 256 = 4 batches * 64 filters
  int f = tid & 63;
  float s   = s_in[f];
  float a   = fminf(a_in[f], 1.0f);
  float d   = fmaxf(d_in[f], 0.0f);
  float rr  = 1.0f / fmaxf(r_in[f], 1.0f);
  float drr = powf(d, rr);
  const float log_eps = logf(EPS_V);
  const float* row = pooled + (size_t)tid * T_OUTN;
  float*      orow = out    + (size_t)tid * T_OUTN;
  float m = 0.0f;
  for (int t = 0; t < T_OUTN; ++t) {
    float y = fmaxf(row[t], CLAMP_V);
    m = (t == 0) ? y : (1.0f - s) * m + s * y;
    float sm = expf(-a * (log_eps + log1pf(m / EPS_V)));
    orow[t] = powf(y * sm + d, rr) - drr;
  }
}

// ---------------------------------------------------------------------------

extern "C" void kernel_launch(void* const* d_in, const int* in_sizes, int n_in,
                              void* d_out, int out_size, void* d_ws, size_t ws_size,
                              hipStream_t stream) {
  const float* x     = (const float*)d_in[0];
  const float* cf    = (const float*)d_in[1];
  const float* bw    = (const float*)d_in[2];
  const float* pw    = (const float*)d_in[3];
  const float* s_p   = (const float*)d_in[4];
  const float* a_p   = (const float*)d_in[5];
  const float* del_p = (const float*)d_in[6];
  const float* r_p   = (const float*)d_in[7];
  float* out = (float*)d_out;

  // workspace carve-out (~2.42 MB total), 256B-aligned regions
  char* ws = (char*)d_ws;
  size_t o = 0;
  auto carve = [&](size_t bytes) { size_t r = o; o = (o + bytes + 255) & ~(size_t)255; return r; };
  _Float16* xp   = (_Float16*)(ws + carve((size_t)NB * XPS * sizeof(_Float16)));
  _Float16* aswz = (_Float16*)(ws + carve((size_t)8 * NKC * 32 * 16 * sizeof(_Float16)));
  float*    win  = (float*)   (ws + carve((size_t)NF * FS * sizeof(float)));
  float*    pool = (float*)   (ws + carve((size_t)NB * NF * T_OUTN * sizeof(float)));
  (void)ws_size; (void)in_sizes; (void)n_in; (void)out_size;

  k_build_xp  <<<(NB * XPS + 255) / 256, 256, 0, stream>>>(x, xp);
  k_build_filt<<<(8 * NKC * 512 + 255) / 256, 256, 0, stream>>>(cf, bw, aswz);
  k_build_win <<<(NF * FS + 255) / 256, 256, 0, stream>>>(pw, win);
  k_zero      <<<(NB * NF * T_OUTN + 255) / 256, 256, 0, stream>>>(pool, NB * NF * T_OUTN);

  k_conv_pool<<<dim3(T_IN / 16, NB), 256, 0, stream>>>(xp, aswz, win, pool);

  k_pcen<<<1, 256, 0, stream>>>(pool, s_p, a_p, del_p, r_p, out);
}
